// MultiHeadAttention_44521630990434
// MI455X (gfx1250) — matrix operand order repaired and checked by reading, not compiled, not run on hardware
//
#include <hip/hip_runtime.h>
#include <hip/hip_bf16.h>

typedef __bf16 bf16_t;
typedef __attribute__((ext_vector_type(16))) __bf16 v16bf;
typedef __attribute__((ext_vector_type(8)))  __bf16 v8bf;
typedef __attribute__((ext_vector_type(8)))  float  v8f;

constexpr int Bn  = 2;
constexpr int Sn  = 2048;
constexpr int Dn  = 1024;
constexpr int Hn  = 16;
constexpr int DKn = 64;
constexpr int Mrows = Bn * Sn;     // 4096 rows for the projection GEMMs
constexpr int RS   = 2052;         // LDS row stride (floats): 2048 + 4 pad
                                   // row stride bytes = 8208 = 16*513 (b128-aligned),
                                   // 8*RS mod 64 banks != 0 -> lanes 16-31 hit
                                   // different banks than lanes 0-15.

// ---------------------------------------------------------------------------
// Fragment loaders for CDNA5 WMMA (wave32); A/B use contiguous k order (same in both), not the ISA table.
// A (16x32 bf16, MxK): lane = m + 16*h ; element j holds K = j + 16*h
// B (32x16 bf16, KxN): lane = n + 16*h ; element j holds K = j + 16*h
// C/D (16x16 f32):     lane = n + 16*h ; element i holds M = i + 8*h
// Both operands are loaded "K-contiguous": A row-major MxK, B as Bt row-major NxK.
// ---------------------------------------------------------------------------
static __device__ inline v16bf frag_a_bf(const bf16_t* __restrict__ A, int lda,
                                         int row, int k0, int h) {
  const bf16_t* p = A + (size_t)row * lda + k0;
  v8bf lo = *reinterpret_cast<const v8bf*>(p + 16 * h);
  v8bf hi = *reinterpret_cast<const v8bf*>(p + 16 * h + 8);
  return __builtin_shufflevector(lo, hi, 0,1,2,3,4,5,6,7,8,9,10,11,12,13,14,15);
}

static __device__ inline v16bf frag_b_bf(const bf16_t* __restrict__ Bt, int ldb,
                                         int col, int k0, int h) {
  const bf16_t* p = Bt + (size_t)col * ldb + k0 + 16 * h;
  v8bf lo = *reinterpret_cast<const v8bf*>(p);
  v8bf hi = *reinterpret_cast<const v8bf*>(p + 8);
  return __builtin_shufflevector(lo, hi, 0,1,2,3,4,5,6,7,8,9,10,11,12,13,14,15);
}

static __device__ inline v8f wmma_bf16(v16bf a, v16bf b, v8f c) {
  return __builtin_amdgcn_wmma_f32_16x16x32_bf16(false, a, false, b,
                                                 (short)0, c, false, false);
}

// ---------------------------------------------------------------------------
// fp32 -> bf16 elementwise conversion
// ---------------------------------------------------------------------------
__global__ void k_cvt_bf16(const float* __restrict__ x, bf16_t* __restrict__ y, int n) {
  int i = blockIdx.x * blockDim.x + threadIdx.x;
  if (i < n) y[i] = (bf16_t)x[i];
}

// ---------------------------------------------------------------------------
// QKV projection: dst = X @ W^T + bias (then *scale), reshaped to head layout.
// vmode 0: dst[(b*H+h)*S + s][dk]  (q, k)     vmode 1: dst[(b*H+h)*DK + dk][s]  (v^T)
// One wave per 16x16 tile, K-loop over D in steps of 32.
// ---------------------------------------------------------------------------
__global__ void k_proj(const bf16_t* __restrict__ X, const bf16_t* __restrict__ W,
                       const float* __restrict__ bias, bf16_t* __restrict__ dst,
                       int vmode, float scale) {
  int wid  = (blockIdx.x * blockDim.x + threadIdx.x) >> 5;
  int lane = threadIdx.x & 31;
  int h    = lane >> 4;
  int l16  = lane & 15;
  const int tilesN = Dn / 16;                 // 64
  int tM = (wid / tilesN) * 16;
  int tN = (wid % tilesN) * 16;

  v8f c = {};
  for (int k0 = 0; k0 < Dn; k0 += 32) {
    v16bf a = frag_a_bf(X, Dn, tM + l16, k0, h);
    v16bf b = frag_b_bf(W, Dn, tN + l16, k0, h);
    c = wmma_bf16(a, b, c);
  }

  int col = tN + l16;
  int hh  = col / DKn, dk = col % DKn;
  float bb = bias[col];
#pragma unroll
  for (int i = 0; i < 8; ++i) {
    int row = tM + i + 8 * h;
    int b_ = row / Sn, s = row % Sn;
    float val = (c[i] + bb) * scale;
    size_t idx = (vmode == 0)
        ? ((size_t)(b_ * Hn + hh) * Sn + s) * DKn + dk
        : ((size_t)(b_ * Hn + hh) * DKn + dk) * Sn + s;
    dst[idx] = (bf16_t)val;
  }
}

// ---------------------------------------------------------------------------
// Fused causal attention over a 32-row strip, score strip resident in LDS.
//   Phase 1: scores = q_scaled @ k^T (WMMA) -> LDS, causal-masked, only the
//            ctiles = (m0+32)/16 column tiles that can be unmasked.
//   Phase 2: row softmax in LDS; normalized probs written back to LDS and
//            streamed ONCE to the global attn output with NON-TEMPORAL stores
//            (536 MB stream-out must not evict the L2-resident WMMA operands).
//   Phase 3: out = attn @ V with A-fragments converted f32->bf16 from LDS,
//            B = V^T (K-contiguous), K-loop bounded at the causal diagonal.
// One 256-thread (8-wave) workgroup per strip; dynamic LDS = 32*RS*4 bytes.
// ---------------------------------------------------------------------------
__global__ void k_attn(const bf16_t* __restrict__ qh, const bf16_t* __restrict__ kh,
                       const bf16_t* __restrict__ vT, float* __restrict__ attn,
                       bf16_t* __restrict__ outh) {
  extern __shared__ float smem[];            // [32][RS]
  int wave = threadIdx.x >> 5;               // 0..7
  int lane = threadIdx.x & 31;
  int h    = lane >> 4;
  int l16  = lane & 15;

  int strip = blockIdx.x;                    // 0 .. Bn*Hn*(Sn/32)-1
  int bh = strip / (Sn / 32);
  int m0 = (strip % (Sn / 32)) * 32;         // first query row of strip

  const bf16_t* q  = qh + (size_t)bh * Sn * DKn;
  const bf16_t* k  = kh + (size_t)bh * Sn * DKn;
  const bf16_t* vt = vT + (size_t)bh * DKn * Sn;
  float* abase = attn + (size_t)bh * Sn * Sn;

  // ---- Phase 1: masked scores into LDS ----
  int ctiles = (m0 + 32) >> 4;               // 16-col tiles with any unmasked entry
  int ntiles = 2 * ctiles;                   // x2 row-tiles in the strip
  for (int tt = wave; tt < ntiles; tt += 8) {
    int rt = tt & 1;
    int ct = tt >> 1;
    v8f c = {};
#pragma unroll
    for (int k0 = 0; k0 < DKn; k0 += 32) {
      v16bf a = frag_a_bf(q, DKn, m0 + rt * 16 + l16, k0, h);
      v16bf b = frag_b_bf(k, DKn, ct * 16 + l16, k0, h);
      c = wmma_bf16(a, b, c);
    }
    int cc = ct * 16 + l16;
#pragma unroll
    for (int i = 0; i < 8; ++i) {
      int rloc = rt * 16 + i + 8 * h;        // local row 0..31
      smem[rloc * RS + cc] = (cc > m0 + rloc) ? -1e9f : c[i];
    }
  }
  __syncthreads();

  // ---- Phase 2: softmax per row (4 rows per wave), LDS-resident ----
  int tlim = (m0 + 32) >> 5;                 // valid 32-wide chunks in LDS
  for (int lr = wave * 4; lr < wave * 4 + 4; ++lr) {
    int i = m0 + lr;
    float* ls = smem + lr * RS;
    float mx = -3.0e38f;
    for (int t = 0; t < tlim; ++t) mx = fmaxf(mx, ls[lane + 32 * t]);
#pragma unroll
    for (int off = 16; off; off >>= 1) mx = fmaxf(mx, __shfl_xor(mx, off, 32));
    float sum = 0.f;
    for (int t = 0; t < tlim; ++t) {
      float e = __expf(ls[lane + 32 * t] - mx);
      ls[lane + 32 * t] = e;
      sum += e;
    }
#pragma unroll
    for (int off = 16; off; off >>= 1) sum += __shfl_xor(sum, off, 32);
    float inv = 1.0f / sum;
    float* grow = abase + (size_t)i * Sn;
    for (int t = 0; t < tlim; ++t) {
      float p = ls[lane + 32 * t] * inv;
      ls[lane + 32 * t] = p;                                 // for phase 3
      __builtin_nontemporal_store(p, grow + lane + 32 * t);  // attn stream-out
    }
    for (int t = tlim; t < Sn / 32; ++t)
      __builtin_nontemporal_store(0.f, grow + lane + 32 * t);
  }
  __syncthreads();

  // ---- Phase 3: out = attn @ V ; one 16x16 tile per wave (2 x 4 tiles) ----
  {
    int rt = wave >> 2;                      // 0..1  (row tile)
    int nt = wave & 3;                       // 0..3  (dk tile)
    int kmax = m0 + 32;                      // causal bound (multiple of 32)
    v8f c = {};
    for (int k0 = 0; k0 < kmax; k0 += 32) {
      const float* p = smem + (rt * 16 + l16) * RS + k0;
      v16bf a;
#pragma unroll
      for (int j = 0; j < 8; ++j) a[j] = (bf16_t)p[16 * h + j];
#pragma unroll
      for (int j = 0; j < 8; ++j) a[8 + j] = (bf16_t)p[16 * h + 8 + j];
      v16bf b = frag_b_bf(vt, Sn, nt * 16 + l16, k0, h);
      c = wmma_bf16(a, b, c);
    }
    int b_ = bh / Hn, hh = bh % Hn;
    int dk = nt * 16 + l16;
#pragma unroll
    for (int i = 0; i < 8; ++i) {
      int s = m0 + rt * 16 + i + 8 * h;
      outh[((size_t)b_ * Sn + s) * Dn + hh * DKn + dk] = (bf16_t)c[i];
    }
  }
}

// ---------------------------------------------------------------------------
// Final projection: out = out_heads @ WO^T + bO  (fp32 result to d_out)
// ---------------------------------------------------------------------------
__global__ void k_outproj(const bf16_t* __restrict__ Xh, const bf16_t* __restrict__ Wo,
                          const float* __restrict__ bias, float* __restrict__ out) {
  int wid  = (blockIdx.x * blockDim.x + threadIdx.x) >> 5;
  int lane = threadIdx.x & 31;
  int h    = lane >> 4;
  int l16  = lane & 15;
  const int tilesN = Dn / 16;
  int tM = (wid / tilesN) * 16;
  int tN = (wid % tilesN) * 16;

  v8f c = {};
  for (int k0 = 0; k0 < Dn; k0 += 32) {
    v16bf a = frag_a_bf(Xh, Dn, tM + l16, k0, h);
    v16bf b = frag_b_bf(Wo, Dn, tN + l16, k0, h);
    c = wmma_bf16(a, b, c);
  }

  int col = tN + l16;
  float bb = bias[col];
#pragma unroll
  for (int i = 0; i < 8; ++i) {
    int row = tM + i + 8 * h;
    out[(size_t)row * Dn + col] = c[i] + bb;
  }
}

// ---------------------------------------------------------------------------
extern "C" void kernel_launch(void* const* d_in, const int* in_sizes, int n_in,
                              void* d_out, int out_size, void* d_ws, size_t ws_size,
                              hipStream_t stream) {
  const float* Q  = (const float*)d_in[0];
  const float* K  = (const float*)d_in[1];
  const float* V  = (const float*)d_in[2];
  // d_in[3] = mask (bool causal tril) — causality is implemented directly.
  const float* WQ = (const float*)d_in[4];  const float* bQ = (const float*)d_in[5];
  const float* WK = (const float*)d_in[6];  const float* bK = (const float*)d_in[7];
  const float* WV = (const float*)d_in[8];  const float* bV = (const float*)d_in[9];
  const float* WO = (const float*)d_in[10]; const float* bO = (const float*)d_in[11];

  float* out  = (float*)d_out;                      // (B,S,D) fp32
  float* attn = out + (size_t)Bn * Sn * Dn;         // (B,H,S,S) fp32

  // workspace carve-up (~64 MiB total)
  char* ws = (char*)d_ws;
  size_t off = 0;
  auto carve = [&](size_t bytes) {
    void* p = ws + off;
    off = (off + bytes + 255) & ~(size_t)255;
    return p;
  };
  const size_t nIn = (size_t)Mrows * Dn;            // 4194304
  const size_t nW  = (size_t)Dn * Dn;               // 1048576
  bf16_t* Qb  = (bf16_t*)carve(nIn * 2);
  bf16_t* Kb  = (bf16_t*)carve(nIn * 2);
  bf16_t* Vb  = (bf16_t*)carve(nIn * 2);
  bf16_t* Wqb = (bf16_t*)carve(nW * 2);
  bf16_t* Wkb = (bf16_t*)carve(nW * 2);
  bf16_t* Wvb = (bf16_t*)carve(nW * 2);
  bf16_t* Wob = (bf16_t*)carve(nW * 2);
  bf16_t* qh  = (bf16_t*)carve((size_t)Bn * Hn * Sn * DKn * 2);
  bf16_t* kh  = (bf16_t*)carve((size_t)Bn * Hn * Sn * DKn * 2);
  bf16_t* vT  = (bf16_t*)carve((size_t)Bn * Hn * DKn * Sn * 2);
  bf16_t* outh= (bf16_t*)carve(nIn * 2);
  (void)ws_size; (void)in_sizes; (void)n_in; (void)out_size;

  dim3 blk(256);

  // 1) fp32 -> bf16 staging
  k_cvt_bf16<<<(int)((nIn + 255) / 256), blk, 0, stream>>>(Q,  Qb,  (int)nIn);
  k_cvt_bf16<<<(int)((nIn + 255) / 256), blk, 0, stream>>>(K,  Kb,  (int)nIn);
  k_cvt_bf16<<<(int)((nIn + 255) / 256), blk, 0, stream>>>(V,  Vb,  (int)nIn);
  k_cvt_bf16<<<(int)((nW  + 255) / 256), blk, 0, stream>>>(WQ, Wqb, (int)nW);
  k_cvt_bf16<<<(int)((nW  + 255) / 256), blk, 0, stream>>>(WK, Wkb, (int)nW);
  k_cvt_bf16<<<(int)((nW  + 255) / 256), blk, 0, stream>>>(WV, Wvb, (int)nW);
  k_cvt_bf16<<<(int)((nW  + 255) / 256), blk, 0, stream>>>(WO, Wob, (int)nW);

  // 2) projections (q pre-scaled by 1/sqrt(DK) = 0.125)
  const int projWaves = (Mrows / 16) * (Dn / 16);   // 16384
  k_proj<<<projWaves * 32 / 256, blk, 0, stream>>>(Qb, Wqb, bQ, qh, 0, 0.125f);
  k_proj<<<projWaves * 32 / 256, blk, 0, stream>>>(Kb, Wkb, bK, kh, 0, 1.0f);
  k_proj<<<projWaves * 32 / 256, blk, 0, stream>>>(Vb, Wvb, bV, vT, 1, 1.0f);

  // 3) fused causal scores + softmax + attn@V, one workgroup per 32-row strip
  const int strips = Bn * Hn * (Sn / 32);           // 2048
  const size_t ldsBytes = (size_t)32 * RS * sizeof(float);  // 262656 B
  k_attn<<<strips, blk, ldsBytes, stream>>>(qh, kh, vT, attn, outh);

  // 4) output projection
  k_outproj<<<projWaves * 32 / 256, blk, 0, stream>>>(outh, Wob, bO, out);
}
